// BiLevelRoutingDeformableAttention_1700807049485
// MI455X (gfx1250) — compile-verified, matrix operand order checked by reading, not run on hardware
//
#include <hip/hip_runtime.h>

typedef __attribute__((ext_vector_type(16))) __bf16 bf16x16;
typedef __attribute__((ext_vector_type(8)))  __bf16 bf16x8;
typedef __attribute__((ext_vector_type(8)))  float  f32x8;
typedef __attribute__((ext_vector_type(4)))  unsigned int u32x4;
typedef __attribute__((ext_vector_type(8)))  int i32x8;
typedef __attribute__((ext_vector_type(4)))  int i32x4;

#define NBATCH 2
#define CCH    128
#define HH     352
#define WWD    352
#define HWSZ   (HH * WWD)          // 123904
#define PIX    (NBATCH * HWSZ)     // 247808
#define NWIN   11
#define P2     121
#define SCALEQ 0.08838834764831845f  // 128^-0.5
#define BNRS   0.9999975000093750f   // rsqrt(1 + 5e-6)

// ---------------------------------------------------------------------------
// helpers
// ---------------------------------------------------------------------------
__device__ __forceinline__ bf16x16 ldfrag(const __bf16* p) {
    // CDNA5 16-bit A/B fragment: per lane, VGPRs 0..3 hold K = kb..kb+7,
    // VGPRs 4..7 hold K = kb+16..kb+23 (kb = (lane>=16)*8, handled by caller).
    bf16x8 lo = *(const bf16x8*)p;
    bf16x8 hi = *(const bf16x8*)(p + 16);
    bf16x16 r;
#pragma unroll
    for (int e = 0; e < 8; ++e) { r[e] = lo[e]; r[8 + e] = hi[e]; }
    return r;
}

__device__ __forceinline__ f32x8 wmma_bf16(bf16x16 a, bf16x16 b, f32x8 c) {
    return __builtin_amdgcn_wmma_f32_16x16x32_bf16(false, a, false, b,
                                                   (short)0, c, false, false);
}

__device__ __forceinline__ float silu(float x) {
    return x / (1.f + __expf(-x));
}

// TDM: 2D tile load global->LDS, bf16 elements (data_size=2B).
// tensor viewed as rows of `strideElems` bf16; copies tileW x tileH into LDS
// contiguously. Descriptor per CDNA5 ISA ch.8 (count=1, type=2).
__device__ __forceinline__ void tdm_load_2d(unsigned lds_off, const void* gptr,
                                            unsigned tileW, unsigned tileH,
                                            unsigned strideElems) {
    unsigned long long ga = (unsigned long long)gptr;
    u32x4 g0;
    g0[0] = 1u;                                            // count=1 (valid)
    g0[1] = lds_off;                                       // lds_addr
    g0[2] = (unsigned)(ga & 0xFFFFFFFFu);                  // global_addr lo
    g0[3] = (unsigned)((ga >> 32) & 0x01FFFFFFu) | (2u << 30);  // hi + type=2
    i32x8 g1;
    g1[0] = (int)(1u << 16);                               // data_size=1 (2B)
    g1[1] = (int)((strideElems & 0xFFFFu) << 16);          // tensor_dim0 lo16
    g1[2] = (int)((strideElems >> 16) | ((tileH & 0xFFFFu) << 16)); // dim0hi|dim1lo
    g1[3] = (int)((tileH >> 16) | (tileW << 16));          // dim1 hi | tile_dim0
    g1[4] = (int)(tileH & 0xFFFFu);                        // tile_dim1 (tile_dim2=0)
    g1[5] = (int)strideElems;                              // tensor_dim0_stride lo32
    g1[6] = 0;                                             // stride hi / dim1_stride
    g1[7] = 0;
    i32x4 z = {0, 0, 0, 0};
#if defined(__clang_major__) && (__clang_major__ >= 23)
    i32x8 z8 = {0, 0, 0, 0, 0, 0, 0, 0};
    __builtin_amdgcn_tensor_load_to_lds(g0, g1, z, z, z8, 0);
#else
    __builtin_amdgcn_tensor_load_to_lds(g0, g1, z, z, 0);
#endif
}

__device__ __forceinline__ unsigned lds_offset_of(const void* p) {
    // ISA 10.2: LDS aperture addresses map to LDS via addr[31:0].
    return (unsigned)(unsigned long long)(uintptr_t)p;
}

// ---------------------------------------------------------------------------
// K0: cast 1x1-conv weights to bf16 (layout [cout][cin], identical to OIHW 1x1)
// ---------------------------------------------------------------------------
__global__ __launch_bounds__(256) void cast_weights_kernel(
    const float* __restrict__ qw, const float* __restrict__ kvw,
    __bf16* __restrict__ qwb, __bf16* __restrict__ kvwb) {
    int i = blockIdx.x * 256 + threadIdx.x;
    if (i < CCH * CCH) qwb[i] = (__bf16)qw[i];
    if (i < 2 * CCH * CCH) kvwb[i] = (__bf16)kvw[i];
}

// ---------------------------------------------------------------------------
// K1: x NCHW f32 -> NHWC bf16
// ---------------------------------------------------------------------------
__global__ __launch_bounds__(256) void cast_x_kernel(
    const float* __restrict__ x, __bf16* __restrict__ xb) {
    size_t i = (size_t)blockIdx.x * 256 + threadIdx.x;
    if (i >= (size_t)PIX * CCH) return;
    int w = (int)(i % WWD);
    int h = (int)((i / WWD) % HH);
    int c = (int)((i / HWSZ) % CCH);
    int n = (int)(i / ((size_t)CCH * HWSZ));
    xb[((size_t)n * HWSZ + (size_t)h * WWD + w) * CCH + c] = (__bf16)x[i];
}

// ---------------------------------------------------------------------------
// K2/K4: GEMM  Y = silu(A[M,128] * W^T[cout,128] + b)   (WMMA bf16->f32)
// block = 256 threads (8 waves); tile = 128 rows x 128 cols; K = 128 staged
// into LDS by the Tensor Data Mover (wave 0 issues, TENSORcnt-tracked).
// ---------------------------------------------------------------------------
__global__ __launch_bounds__(256) void gemm_bias_silu_kernel(
    const __bf16* __restrict__ A, const __bf16* __restrict__ Wt,
    const float* __restrict__ bias, __bf16* __restrict__ outB,
    float* __restrict__ outF, int coutTotal) {
    __shared__ __bf16 sA[128 * 128];
    __shared__ __bf16 sW[128 * 128];
    const int tid = threadIdx.x;
    const size_t rowBase = (size_t)blockIdx.x * 128;
    const int colBase = blockIdx.y * 128;

    // L2 warm-up for the next M tile (global_prefetch_b8)
    __builtin_prefetch(A + (rowBase + 128) * CCH + (size_t)tid * 64, 0, 1);

    if (tid < 32) {  // one wave drives the TDM; EXEC is ignored by TENSOR ops
        tdm_load_2d(lds_offset_of(sA), A + rowBase * CCH, 128u, 128u, 128u);
        tdm_load_2d(lds_offset_of(sW), Wt + (size_t)colBase * CCH, 128u, 128u,
                    128u);
        __builtin_amdgcn_s_wait_tensorcnt(0);
    }
    __syncthreads();

    const int wave = tid >> 5, lane = tid & 31;
    const int wm = wave & 3;       // 4 wave-rows of 32
    const int wn = wave >> 2;      // 2 wave-cols of 64
    const int l16 = lane & 15;
    const int hi = lane >> 4;
    const int khalf = hi * 8;

    f32x8 acc[2][4];
#pragma unroll
    for (int i = 0; i < 2; ++i)
#pragma unroll
        for (int j = 0; j < 4; ++j) acc[i][j] = {};

#pragma unroll
    for (int kt = 0; kt < 4; ++kt) {
        const int kb = kt * 32 + khalf;
        bf16x16 a[2], b[4];
#pragma unroll
        for (int i = 0; i < 2; ++i)
            a[i] = ldfrag(sA + (wm * 32 + i * 16 + l16) * 128 + kb);
#pragma unroll
        for (int j = 0; j < 4; ++j)
            b[j] = ldfrag(sW + (wn * 64 + j * 16 + l16) * 128 + kb);
#pragma unroll
        for (int i = 0; i < 2; ++i)
#pragma unroll
            for (int j = 0; j < 4; ++j)
                acc[i][j] = wmma_bf16(a[i], b[j], acc[i][j]);
    }

#pragma unroll
    for (int i = 0; i < 2; ++i)
#pragma unroll
        for (int j = 0; j < 4; ++j) {
            const int col = colBase + wn * 64 + j * 16 + l16;
            const float bv = bias[col];
#pragma unroll
            for (int v = 0; v < 8; ++v) {
                const size_t row = rowBase + wm * 32 + i * 16 + v + hi * 8;
                const float y = silu(acc[i][j][v] + bv);
                if (outB) outB[row * coutTotal + col] = (__bf16)y;
                if (outF) outF[row * coutTotal + col] = y;
            }
        }
}

// ---------------------------------------------------------------------------
// K3: fused offset (dw9x9 + BN + pw->2) + clamp + bilinear sample -> xs bf16
// grid (22,22,N), block (16,16)
// ---------------------------------------------------------------------------
__global__ __launch_bounds__(256) void offset_sample_kernel(
    const __bf16* __restrict__ qb, const __bf16* __restrict__ xb,
    const float* __restrict__ dww, const float* __restrict__ dwb,
    const float* __restrict__ bng, const float* __restrict__ bnb,
    const float* __restrict__ pww, __bf16* __restrict__ xs) {
    __shared__ float wgt[81];
    __shared__ __bf16 tile[24][24];
    const int n = blockIdx.z;
    const int h0 = blockIdx.y * 16, w0 = blockIdx.x * 16;
    const int tx = threadIdx.x, ty = threadIdx.y;
    const int tid = ty * 16 + tx;

    float o0 = 0.f, o1 = 0.f;
    for (int c = 0; c < CCH; ++c) {
        for (int i = tid; i < 576; i += 256) {
            const int r = i / 24, cc = i % 24;
            const int h = h0 - 4 + r, w = w0 - 4 + cc;
            __bf16 v = (__bf16)0.f;
            if (h >= 0 && h < HH && w >= 0 && w < WWD)
                v = qb[((size_t)n * HWSZ + (size_t)h * WWD + w) * CCH + c];
            tile[r][cc] = v;
        }
        if (tid < 81) wgt[tid] = dww[c * 81 + tid];
        __syncthreads();
        float acc = 0.f;
#pragma unroll 3
        for (int dy = 0; dy < 9; ++dy)
#pragma unroll 3
            for (int dx = 0; dx < 9; ++dx)
                acc += wgt[dy * 9 + dx] * (float)tile[ty + dy][tx + dx];
        const float t = (acc + dwb[c]) * (bng[c] * BNRS) + bnb[c];
        o0 += pww[c] * t;
        o1 += pww[CCH + c] * t;
        __syncthreads();
    }

    const int h = h0 + ty, w = w0 + tx;
    const float ry = (0.5f + (float)h) / (float)(HH - 1) * 2.f - 1.f;
    const float rx = (0.5f + (float)w) / (float)(WWD - 1) * 2.f - 1.f;
    const float py = fminf(fmaxf(o0 + ry, -0.12f), 0.12f);
    const float px = fminf(fmaxf(o1 + rx, -0.12f), 0.12f);
    const float gx = (px + 1.f) * 0.5f * (float)(WWD - 1);
    const float gy = (py + 1.f) * 0.5f * (float)(HH - 1);
    const float x0f = floorf(gx), y0f = floorf(gy);
    const float wx = gx - x0f, wy = gy - y0f;
    int x0 = min(max((int)x0f, 0), WWD - 1);
    int x1 = min(x0 + 1, WWD - 1);
    int y0 = min(max((int)y0f, 0), HH - 1);
    int y1 = min(y0 + 1, HH - 1);
    const float w00 = (1.f - wx) * (1.f - wy), w01 = wx * (1.f - wy);
    const float w10 = (1.f - wx) * wy, w11 = wx * wy;
    const __bf16* p00 = xb + ((size_t)n * HWSZ + (size_t)y0 * WWD + x0) * CCH;
    const __bf16* p01 = xb + ((size_t)n * HWSZ + (size_t)y0 * WWD + x1) * CCH;
    const __bf16* p10 = xb + ((size_t)n * HWSZ + (size_t)y1 * WWD + x0) * CCH;
    const __bf16* p11 = xb + ((size_t)n * HWSZ + (size_t)y1 * WWD + x1) * CCH;
    __bf16* dst = xs + ((size_t)n * HWSZ + (size_t)h * WWD + w) * CCH;
#pragma unroll 8
    for (int c = 0; c < CCH; ++c)
        dst[c] = (__bf16)(w00 * (float)p00[c] + w01 * (float)p01[c] +
                          w10 * (float)p10[c] + w11 * (float)p11[c]);
}

// ---------------------------------------------------------------------------
// K5: per-window means of q (bf16) and k-half of kv (f32). grid N*121, 256 thr
// ---------------------------------------------------------------------------
__global__ __launch_bounds__(256) void window_stats_kernel(
    const __bf16* __restrict__ qb, const float* __restrict__ kv,
    float* __restrict__ qwin, float* __restrict__ kwin) {
    const int blk = blockIdx.x, n = blk / P2, p = blk % P2;
    const int a1 = p / NWIN, a2 = p % NWIN;
    const int tid = threadIdx.x;
    float s = 0.f;
    if (tid < CCH) {
        const int c = tid;
        for (int r = 0; r < 1024; ++r) {
            const int h = a1 * 32 + (r >> 5), w = a2 * 32 + (r & 31);
            s += (float)qb[((size_t)n * HWSZ + (size_t)h * WWD + w) * CCH + c];
        }
        qwin[((size_t)n * P2 + p) * CCH + c] = s * (1.f / 1024.f);
    } else {
        const int c = tid - CCH;
        for (int r = 0; r < 1024; ++r) {
            const int h = a1 * 32 + (r >> 5), w = a2 * 32 + (r & 31);
            s += kv[((size_t)n * HWSZ + (size_t)h * WWD + w) * (2 * CCH) + c];
        }
        kwin[((size_t)n * P2 + p) * CCH + c] = s * (1.f / 1024.f);
    }
}

// ---------------------------------------------------------------------------
// K6: routing logits + top-4. grid N*121, block 128
// ---------------------------------------------------------------------------
__global__ __launch_bounds__(128) void routing_kernel(
    const float* __restrict__ qwin, const float* __restrict__ kwin,
    int* __restrict__ topi) {
    __shared__ float logit[P2];
    const int n = blockIdx.x / P2, p = blockIdx.x % P2;
    const int t = threadIdx.x;
    if (t < P2) {
        const float* qp = qwin + ((size_t)n * P2 + p) * CCH;
        const float* kp = kwin + ((size_t)n * P2 + t) * CCH;
        float s = 0.f;
        for (int c = 0; c < CCH; ++c) s += qp[c] * kp[c];
        logit[t] = s * SCALEQ;
    }
    __syncthreads();
    if (t == 0) {
        unsigned long long u0 = 0ull, u1 = 0ull;
        for (int k = 0; k < 4; ++k) {
            float best = -3.0e38f;
            int bi = 0;
            for (int j = 0; j < P2; ++j) {
                const bool used = (j < 64) ? ((u0 >> j) & 1ull)
                                           : ((u1 >> (j - 64)) & 1ull);
                if (!used && logit[j] > best) { best = logit[j]; bi = j; }
            }
            if (bi < 64) u0 |= 1ull << bi; else u1 |= 1ull << (bi - 64);
            topi[((size_t)n * P2 + p) * 4 + k] = bi;
        }
    }
}

// ---------------------------------------------------------------------------
// K7: kvd = 8x8 max-pool of kv windows -> bf16 (N,121,16,256). grid N*121
// ---------------------------------------------------------------------------
__global__ __launch_bounds__(256) void kvd_max_kernel(
    const float* __restrict__ kv, __bf16* __restrict__ kvd) {
    const int blk = blockIdx.x, n = blk / P2, p = blk % P2;
    const int a1 = p / NWIN, a2 = p % NWIN;
    const int c = threadIdx.x;  // 0..255
    for (int s = 0; s < 16; ++s) {
        const int i1 = s >> 2, j1 = s & 3;
        float m = -3.0e38f;
        for (int i2 = 0; i2 < 8; ++i2)
            for (int j2 = 0; j2 < 8; ++j2) {
                const int h = a1 * 32 + i1 * 8 + i2;
                const int w = a2 * 32 + j1 * 8 + j2;
                m = fmaxf(m, kv[((size_t)n * HWSZ + (size_t)h * WWD + w) * 256 + c]);
            }
        kvd[(((size_t)n * P2 + p) * 16 + s) * 256 + c] = (__bf16)m;
    }
}

// ---------------------------------------------------------------------------
// K8: windowed multi-head attention (WMMA). grid N*121, 256 threads = 8 waves.
// Top-k kv gather done by TDM: 4 contiguous 16x256 bf16 tiles -> LDS.
// ---------------------------------------------------------------------------
__global__ __launch_bounds__(256) void attention_kernel(
    const __bf16* __restrict__ qb, const __bf16* __restrict__ kvd,
    const int* __restrict__ topi, float* __restrict__ attn_out) {
    __shared__ __bf16 kvsel[64][256];     // 32 KB  gathered k|v rows
    __shared__ __bf16 vT[4][32][64];      // 16 KB  v transposed per head
    __shared__ __bf16 pbuf[8][16][64];    // 16 KB  per-wave P staging
    const int blk = blockIdx.x, n = blk / P2, p = blk % P2;
    const int a1 = p / NWIN, a2 = p % NWIN;
    const int tid = threadIdx.x;

    if (tid < 32) {  // wave 0 issues the 4 TDM tile loads (16 rows x 512 B)
        const int* ti = topi + ((size_t)n * P2 + p) * 4;
#pragma unroll
        for (int slot = 0; slot < 4; ++slot) {
            const int widx = ti[slot];
            tdm_load_2d(lds_offset_of(&kvsel[slot * 16][0]),
                        kvd + (((size_t)n * P2 + widx) * 16) * 256, 256u, 16u,
                        256u);
        }
        __builtin_amdgcn_s_wait_tensorcnt(0);
    }
    __syncthreads();
    for (int i = tid; i < 4 * 32 * 64; i += 256) {
        const int h = i >> 11, d = (i >> 6) & 31, j = i & 63;
        vT[h][d][j] = kvsel[j][128 + h * 32 + d];
    }
    __syncthreads();

    const int wave = tid >> 5, lane = tid & 31;
    const int l16 = lane & 15, hi = lane >> 4, khalf = hi * 8;

    for (int unit = wave; unit < 256; unit += 8) {   // 4 heads * 64 m-tiles
        const int head = unit >> 6;
        const int mtile = unit & 63;

        // A fragment: 16 q rows x 32 k (one shot, K=32)
        const int r = mtile * 16 + l16;
        const int h = a1 * 32 + (r >> 5), w = a2 * 32 + (r & 31);
        const bf16x16 afrag = ldfrag(
            qb + ((size_t)n * HWSZ + (size_t)h * WWD + w) * CCH + head * 32 + khalf);

        // S = (q*scale) K^T : 4 N-tiles of 16
        f32x8 s[4];
#pragma unroll
        for (int nt = 0; nt < 4; ++nt) {
            const bf16x16 bfrag =
                ldfrag(&kvsel[nt * 16 + l16][head * 32 + khalf]);
            f32x8 z = {};
            s[nt] = wmma_bf16(afrag, bfrag, z);
        }

        // softmax over 64 cols: rows live per-VGPR, cols across 16-lane halves
        f32x8 pr[4];
#pragma unroll
        for (int v = 0; v < 8; ++v) {
            float m = -3.0e38f;
#pragma unroll
            for (int nt = 0; nt < 4; ++nt) {
                s[nt][v] *= SCALEQ;
                m = fmaxf(m, s[nt][v]);
            }
#pragma unroll
            for (int msk = 1; msk < 16; msk <<= 1)
                m = fmaxf(m, __shfl_xor(m, msk, 32));
            float sum = 0.f;
#pragma unroll
            for (int nt = 0; nt < 4; ++nt) {
                const float e = __expf(s[nt][v] - m);
                pr[nt][v] = e;
                sum += e;
            }
#pragma unroll
            for (int msk = 1; msk < 16; msk <<= 1)
                sum += __shfl_xor(sum, msk, 32);
            const float rs = 1.f / sum;
#pragma unroll
            for (int nt = 0; nt < 4; ++nt) pr[nt][v] *= rs;
        }

        // stage P (C-layout -> A-layout via LDS), wave-private buffer
#pragma unroll
        for (int nt = 0; nt < 4; ++nt)
#pragma unroll
            for (int v = 0; v < 8; ++v)
                pbuf[wave][v + hi * 8][nt * 16 + l16] = (__bf16)pr[nt][v];

        // O = P @ V : M=16,K=64,N=32
        f32x8 o[2];
        o[0] = {}; o[1] = {};
#pragma unroll
        for (int kt = 0; kt < 2; ++kt) {
            const bf16x16 pa = ldfrag(&pbuf[wave][l16][kt * 32 + khalf]);
#pragma unroll
            for (int nt = 0; nt < 2; ++nt) {
                const bf16x16 vb =
                    ldfrag(&vT[head][nt * 16 + l16][kt * 32 + khalf]);
                o[nt] = wmma_bf16(pa, vb, o[nt]);
            }
        }
#pragma unroll
        for (int nt = 0; nt < 2; ++nt)
#pragma unroll
            for (int v = 0; v < 8; ++v) {
                const int row = mtile * 16 + v + hi * 8;
                const int col = head * 32 + nt * 16 + l16;
                attn_out[(((size_t)n * P2 + p) * 1024 + row) * CCH + col] =
                    o[nt][v];
            }
    }
}

// ---------------------------------------------------------------------------
// K9: sec depthwise 3x3 on v-half of kv + add attention output -> NCHW f32
// ---------------------------------------------------------------------------
__global__ __launch_bounds__(256) void final_combine_kernel(
    const float* __restrict__ attn_out, const float* __restrict__ kv,
    const float* __restrict__ secw, const float* __restrict__ secb,
    float* __restrict__ out) {
    size_t idx = (size_t)blockIdx.x * 256 + threadIdx.x;
    if (idx >= (size_t)PIX * CCH) return;
    const int w = (int)(idx % WWD);
    const int h = (int)((idx / WWD) % HH);
    const int c = (int)((idx / HWSZ) % CCH);
    const int n = (int)(idx / ((size_t)CCH * HWSZ));
    float s = secb[c];
#pragma unroll
    for (int dy = 0; dy < 3; ++dy)
#pragma unroll
        for (int dx = 0; dx < 3; ++dx) {
            const int hh = h + dy - 1, ww = w + dx - 1;
            if (hh >= 0 && hh < HH && ww >= 0 && ww < WWD)
                s += secw[c * 9 + dy * 3 + dx] *
                     kv[((size_t)n * HWSZ + (size_t)hh * WWD + ww) * 256 + 128 + c];
        }
    const int a1 = h >> 5, a2 = w >> 5;
    const int r = (h & 31) * 32 + (w & 31);
    const int p = a1 * NWIN + a2;
    out[idx] =
        attn_out[(((size_t)n * P2 + p) * 1024 + r) * CCH + c] + s;
}

// ---------------------------------------------------------------------------
// host launch
// ---------------------------------------------------------------------------
extern "C" void kernel_launch(void* const* d_in, const int* in_sizes, int n_in,
                              void* d_out, int out_size, void* d_ws,
                              size_t ws_size, hipStream_t stream) {
    (void)in_sizes; (void)n_in; (void)out_size; (void)ws_size;
    const float* x      = (const float*)d_in[0];
    const float* q_w    = (const float*)d_in[1];
    const float* q_b    = (const float*)d_in[2];
    const float* kv_w   = (const float*)d_in[3];
    const float* kv_b   = (const float*)d_in[4];
    const float* dw_w   = (const float*)d_in[5];
    const float* dw_b   = (const float*)d_in[6];
    const float* bn_g   = (const float*)d_in[7];
    const float* bn_b   = (const float*)d_in[8];
    const float* pw_w   = (const float*)d_in[9];
    const float* sec_w  = (const float*)d_in[10];
    const float* sec_b  = (const float*)d_in[11];
    float* out = (float*)d_out;

    char* base = (char*)d_ws;
    size_t off = 0;
    auto alloc = [&](size_t bytes) -> void* {
        void* p = base + off;
        off = (off + bytes + 255) & ~(size_t)255;
        return p;
    };
    __bf16* x_bf   = (__bf16*)alloc((size_t)PIX * CCH * 2);
    __bf16* q_bf   = (__bf16*)alloc((size_t)PIX * CCH * 2);
    __bf16* xs_bf  = (__bf16*)alloc((size_t)PIX * CCH * 2);
    float*  kv_f   = (float*)alloc((size_t)PIX * 256 * 4);
    __bf16* kvd    = (__bf16*)alloc((size_t)NBATCH * P2 * 16 * 256 * 2);
    float*  qwin   = (float*)alloc((size_t)NBATCH * P2 * CCH * 4);
    float*  kwin   = (float*)alloc((size_t)NBATCH * P2 * CCH * 4);
    int*    topi   = (int*)alloc((size_t)NBATCH * P2 * 4 * 4);
    __bf16* qwb    = (__bf16*)alloc((size_t)CCH * CCH * 2);
    __bf16* kvwb   = (__bf16*)alloc((size_t)2 * CCH * CCH * 2);
    float*  a_out  = (float*)alloc((size_t)NBATCH * P2 * 1024 * CCH * 4);

    cast_weights_kernel<<<128, 256, 0, stream>>>(q_w, kv_w, qwb, kvwb);
    cast_x_kernel<<<(unsigned)(((size_t)PIX * CCH + 255) / 256), 256, 0, stream>>>(
        x, x_bf);
    gemm_bias_silu_kernel<<<dim3(PIX / 128, 1), 256, 0, stream>>>(
        x_bf, qwb, q_b, q_bf, nullptr, CCH);
    offset_sample_kernel<<<dim3(WWD / 16, HH / 16, NBATCH), dim3(16, 16), 0,
                           stream>>>(q_bf, x_bf, dw_w, dw_b, bn_g, bn_b, pw_w,
                                     xs_bf);
    gemm_bias_silu_kernel<<<dim3(PIX / 128, 2), 256, 0, stream>>>(
        xs_bf, kvwb, kv_b, nullptr, kv_f, 2 * CCH);
    window_stats_kernel<<<NBATCH * P2, 256, 0, stream>>>(q_bf, kv_f, qwin, kwin);
    routing_kernel<<<NBATCH * P2, 128, 0, stream>>>(qwin, kwin, topi);
    kvd_max_kernel<<<NBATCH * P2, 256, 0, stream>>>(kv_f, kvd);
    attention_kernel<<<NBATCH * P2, 256, 0, stream>>>(q_bf, kvd, topi, a_out);
    final_combine_kernel<<<(unsigned)(((size_t)PIX * CCH + 255) / 256), 256, 0,
                           stream>>>(a_out, kv_f, sec_w, sec_b, out);
}
